// GraphSAGE_87892210745354
// MI455X (gfx1250) — matrix operand order repaired and checked by reading, not compiled, not run on hardware
//
#include <hip/hip_runtime.h>

#define BM 128
#define BN 128
#define KT 32
#define APITCH 40   // 32 + 8 pad (bf16 elems) -> 80B row pitch, conflict-free
#define WPITCH 40

typedef __attribute__((ext_vector_type(16))) __bf16 v16bf;
typedef __attribute__((ext_vector_type(8)))  __bf16 v8bf;
typedef __attribute__((ext_vector_type(2)))  __bf16 v2bf;
typedef __attribute__((ext_vector_type(8)))  float  v8f;

static __device__ __forceinline__ v16bf cat8(v8bf a, v8bf b) {
  return __builtin_shufflevector(a, b, 0,1,2,3,4,5,6,7,8,9,10,11,12,13,14,15);
}

// ---------------- degree / inverse-degree ----------------
__global__ void deg_kernel(const int* __restrict__ dst, float* __restrict__ deg, int E) {
  int e = blockIdx.x * blockDim.x + threadIdx.x;
  if (e < E) atomicAdd(&deg[dst[e]], 1.0f);
}

__global__ void invdeg_kernel(const float* __restrict__ deg, float* __restrict__ inv, int N) {
  int i = blockIdx.x * blockDim.x + threadIdx.x;
  if (i < N) { float d = deg[i]; inv[i] = d > 0.f ? 1.f / d : 0.f; }
}

// ---------------- edge aggregation: neigh[dst[e],:] += h[src[e],:] ----------------
__global__ __launch_bounds__(256) void agg_kernel(const float* __restrict__ h,
    const int* __restrict__ src, const int* __restrict__ dst,
    float* __restrict__ neigh, int E, int H) {
  int lane = threadIdx.x & 31;
  int gw = blockIdx.x * (blockDim.x >> 5) + (threadIdx.x >> 5);  // one wave per edge
  if (gw >= E) return;
  int s = src[gw], d = dst[gw];
  const float4* hp = (const float4*)(h + (size_t)s * H);
  float* np = neigh + (size_t)d * H;
  int nq = H >> 2;
  for (int i = lane; i < nq; i += 32) {
    float4 v = hp[i];
    atomicAdd(&np[4*i+0], v.x);
    atomicAdd(&np[4*i+1], v.y);
    atomicAdd(&np[4*i+2], v.z);
    atomicAdd(&np[4*i+3], v.w);
  }
}

// ---------------- dual GEMM: C = f(A1)@W1 + g(A2)@W2 (+bias)(+relu) ----------------
// fp32 inputs, split-precision bf16 WMMA (Ah*Bh + Ah*Bl + Al*Bh), fp32 accumulate.
// Register-prefetch pipeline: global loads for step s+1 issue before compute of step s.
__global__ __launch_bounds__(256) void gemm_dual(
    const float* __restrict__ A1, const float* __restrict__ A2,
    const float* __restrict__ rowScaleA2,
    const float* __restrict__ W1, const float* __restrict__ W2,
    const float* __restrict__ bias,
    float* __restrict__ Cout,
    int M, int Nn, int Kh, int reluA, int reluOut)
{
  __shared__ __align__(16) __bf16 Ah[BM * APITCH];
  __shared__ __align__(16) __bf16 Al[BM * APITCH];
  __shared__ __align__(16) __bf16 Wh[BN * WPITCH];  // transposed: [n][k]
  __shared__ __align__(16) __bf16 Wl[BN * WPITCH];

  const int tid  = threadIdx.x;
  const int lane = tid & 31;
  const int wave = tid >> 5;          // 8 waves; wave owns N-cols [16w,16w+16)
  const int half = lane >> 4;
  const int l15  = lane & 15;
  const int m0 = blockIdx.x * BM;
  const int n0 = blockIdx.y * BN;
  const int nsteps = (2 * Kh) / KT;

  // --- per-thread staging coordinates (fixed across steps) ---
  // A tile: BM x KT floats as 2048 float2; thread owns 8: r = idx>>4, k2 = idx&15
  // W tile: KT x BN floats; thread owns 8 pairs: kk2 = idx>>7 (k=2*kk2,2*kk2+1), n = idx&127
  float2 pa[8];
  float2 pw[8];   // .x = W[2kk2][n], .y = W[2kk2+1][n]

  auto load_tile = [&](int s) {
    const int kc = s * KT;
    const bool second = (kc >= Kh);
    const float* __restrict__ Asrc = second ? A2 : A1;
    const float* __restrict__ Wsrc = second ? W2 : W1;
    const int kloc = second ? (kc - Kh) : kc;
#pragma unroll
    for (int it = 0; it < 8; ++it) {
      int idx = tid + it * 256;
      int r = idx >> 4, k2 = idx & 15;
      int grow = m0 + r;
      float2 v = make_float2(0.f, 0.f);
      if (grow < M) {
        v = *(const float2*)&Asrc[(size_t)grow * Kh + kloc + 2 * k2];
        if (reluA) { v.x = v.x > 0.f ? v.x : 0.f; v.y = v.y > 0.f ? v.y : 0.f; }
        if (second && rowScaleA2) { float sc = rowScaleA2[grow]; v.x *= sc; v.y *= sc; }
      }
      pa[it] = v;
    }
#pragma unroll
    for (int it = 0; it < 8; ++it) {
      int idx = tid + it * 256;
      int kk2 = idx >> 7, n = idx & 127;
      const float* wp = &Wsrc[(size_t)(kloc + 2 * kk2) * Nn + (n0 + n)];
      pw[it] = make_float2(wp[0], wp[Nn]);
    }
  };

  auto store_tile = [&]() {
#pragma unroll
    for (int it = 0; it < 8; ++it) {
      int idx = tid + it * 256;
      int r = idx >> 4, k2 = idx & 15;
      float2 v = pa[it];
      __bf16 h0 = (__bf16)v.x, h1 = (__bf16)v.y;
      __bf16 l0 = (__bf16)(v.x - (float)h0), l1 = (__bf16)(v.y - (float)h1);
      *(v2bf*)&Ah[r * APITCH + 2 * k2] = (v2bf){h0, h1};
      *(v2bf*)&Al[r * APITCH + 2 * k2] = (v2bf){l0, l1};
    }
#pragma unroll
    for (int it = 0; it < 8; ++it) {
      int idx = tid + it * 256;
      int kk2 = idx >> 7, n = idx & 127;
      float2 v = pw[it];
      __bf16 h0 = (__bf16)v.x, h1 = (__bf16)v.y;
      __bf16 l0 = (__bf16)(v.x - (float)h0), l1 = (__bf16)(v.y - (float)h1);
      *(v2bf*)&Wh[n * WPITCH + 2 * kk2] = (v2bf){h0, h1};
      *(v2bf*)&Wl[n * WPITCH + 2 * kk2] = (v2bf){l0, l1};
    }
  };

  v8f acc[8];
  v8f zero = {};
#pragma unroll
  for (int t = 0; t < 8; ++t) acc[t] = zero;

  load_tile(0);
  for (int s = 0; s < nsteps; ++s) {
    __syncthreads();          // previous compute done reading LDS
    store_tile();
    __syncthreads();          // LDS tile ready
    if (s + 1 < nsteps) load_tile(s + 1);   // prefetch next tile; latency hides under WMMA

    // B fragment (32x16 bf16): lanes0-15 N=l15,K=0..15 ; lanes16-31 N=l15,K=16..31
    const __bf16* wbh = &Wh[(wave * 16 + l15) * WPITCH + 16 * half];
    const __bf16* wbl = &Wl[(wave * 16 + l15) * WPITCH + 16 * half];
    v16bf bh = cat8(*(const v8bf*)wbh, *(const v8bf*)(wbh + 8));
    v16bf bl = cat8(*(const v8bf*)wbl, *(const v8bf*)(wbl + 8));

#pragma unroll
    for (int t = 0; t < 8; ++t) {
      // A fragment (16x32 bf16): elems 0-15 -> K=16*half..16*half+15
      const __bf16* abh = &Ah[(t * 16 + l15) * APITCH + 16 * half];
      const __bf16* abl = &Al[(t * 16 + l15) * APITCH + 16 * half];
      v16bf ah = cat8(*(const v8bf*)abh, *(const v8bf*)(abh + 8));
      v16bf al = cat8(*(const v8bf*)abl, *(const v8bf*)(abl + 8));
      acc[t] = __builtin_amdgcn_wmma_f32_16x16x32_bf16(false, ah, false, bh, (short)0, acc[t], false, false);
      acc[t] = __builtin_amdgcn_wmma_f32_16x16x32_bf16(false, ah, false, bl, (short)0, acc[t], false, false);
      acc[t] = __builtin_amdgcn_wmma_f32_16x16x32_bf16(false, al, false, bh, (short)0, acc[t], false, false);
    }
  }

  // epilogue: D vgpr i -> M = 8*half + i ; N = l15
  const int ncol = n0 + wave * 16 + l15;
  const float badd = bias ? bias[ncol] : 0.f;
#pragma unroll
  for (int t = 0; t < 8; ++t) {
#pragma unroll
    for (int i = 0; i < 8; ++i) {
      int row = m0 + t * 16 + half * 8 + i;
      if (row < M) {
        float v = acc[t][i] + badd;
        if (reluOut) v = v > 0.f ? v : 0.f;
        Cout[(size_t)row * Nn + ncol] = v;
      }
    }
  }
}

extern "C" void kernel_launch(void* const* d_in, const int* in_sizes, int n_in,
                              void* d_out, int out_size, void* d_ws, size_t ws_size,
                              hipStream_t stream) {
  const int H  = 512;
  const int Cc = 128;
  const int N  = in_sizes[0] / H;   // 50000
  const int E  = in_sizes[1];       // 500000

  const float* feat = (const float*)d_in[0];
  const int*   src  = (const int*)d_in[1];
  const int*   dst  = (const int*)d_in[2];
  const float* Ws   = (const float*)d_in[3];   // [3,H,H]
  const float* Wn   = (const float*)d_in[4];   // [3,H,H]
  const float* bvec = (const float*)d_in[5];   // [3,H]
  const float* Wpp  = (const float*)d_in[6];   // [2H,Cc]

  char* ws = (char*)d_ws;
  auto take = [&](size_t bytes) { char* p = ws; ws += (bytes + 255) & ~(size_t)255; return p; };
  float* deg   = (float*)take((size_t)N * 4);
  float* inv   = (float*)take((size_t)N * 4);
  float* neigh = (float*)take((size_t)N * H * 4);
  float* bufA  = (float*)take((size_t)N * H * 4);
  float* bufB  = (float*)take((size_t)N * H * 4);

  // degrees -> inv_deg
  hipMemsetAsync(deg, 0, (size_t)N * 4, stream);
  deg_kernel<<<(E + 255) / 256, 256, 0, stream>>>(dst, deg, E);
  invdeg_kernel<<<(N + 255) / 256, 256, 0, stream>>>(deg, inv, N);

  dim3 blk(256);
  dim3 ggrid((N + BM - 1) / BM, H / BN);     // 391 x 4
  int aggBlocks = (E + 7) / 8;               // one wave per edge

  // layer 0: h1 = relu(feat@Ws0 + mean@Wn0 + b0)
  hipMemsetAsync(neigh, 0, (size_t)N * H * 4, stream);
  agg_kernel<<<aggBlocks, 256, 0, stream>>>(feat, src, dst, neigh, E, H);
  gemm_dual<<<ggrid, blk, 0, stream>>>(feat, neigh, inv, Ws, Wn, bvec, bufA,
                                       N, H, H, /*reluA=*/0, /*reluOut=*/1);
  // layer 1: h2 = relu(h1@Ws1 + mean@Wn1 + b1)
  hipMemsetAsync(neigh, 0, (size_t)N * H * 4, stream);
  agg_kernel<<<aggBlocks, 256, 0, stream>>>(bufA, src, dst, neigh, E, H);
  gemm_dual<<<ggrid, blk, 0, stream>>>(bufA, neigh, inv,
                                       Ws + (size_t)H * H, Wn + (size_t)H * H, bvec + H, bufB,
                                       N, H, H, 0, 1);
  // layer 2: h3 = h2@Ws2 + mean@Wn2 + b2   (no relu)
  hipMemsetAsync(neigh, 0, (size_t)N * H * 4, stream);
  agg_kernel<<<aggBlocks, 256, 0, stream>>>(bufB, src, dst, neigh, E, H);
  gemm_dual<<<ggrid, blk, 0, stream>>>(bufB, neigh, inv,
                                       Ws + 2 * (size_t)H * H, Wn + 2 * (size_t)H * H, bvec + 2 * H, bufA,
                                       N, H, H, 0, 0);
  // final: out = [relu(h3) | relu(h2)] @ Wpp   (Wpp already stacked [2H,Cc])
  dim3 fgrid((N + BM - 1) / BM, Cc / BN);    // 391 x 1
  gemm_dual<<<fgrid, blk, 0, stream>>>(bufA, bufB, nullptr,
                                       Wpp, Wpp + (size_t)H * Cc, nullptr, (float*)d_out,
                                       N, Cc, H, /*reluA=*/1, /*reluOut=*/0);
}
